// MultiHeadAttention_26577257628291
// MI455X (gfx1250) — matrix operand order repaired and checked by reading, not compiled, not run on hardware
//
#include <hip/hip_runtime.h>

#define HID 1024
#define NH  16
#define HD  64
#define NB  2
#define SQ  2048

typedef __attribute__((ext_vector_type(16))) __bf16 v16bf;
typedef __attribute__((ext_vector_type(8)))  float  v8f;

union FragB { v16bf v; uint4 q[2]; };

__device__ __forceinline__ unsigned short f2bf(float f) {
  unsigned int u = __float_as_uint(f);
  u += 0x7FFFu + ((u >> 16) & 1u);   // round-to-nearest-even
  return (unsigned short)(u >> 16);
}

__device__ __forceinline__ float rmax16(float x) {
  x = fmaxf(x, __shfl_xor(x, 1, 32));
  x = fmaxf(x, __shfl_xor(x, 2, 32));
  x = fmaxf(x, __shfl_xor(x, 4, 32));
  x = fmaxf(x, __shfl_xor(x, 8, 32));
  return x;
}
__device__ __forceinline__ float rsum16(float x) {
  x += __shfl_xor(x, 1, 32);
  x += __shfl_xor(x, 2, 32);
  x += __shfl_xor(x, 4, 32);
  x += __shfl_xor(x, 8, 32);
  return x;
}

// ---------------- fp32 -> bf16 conversion ----------------
__global__ __launch_bounds__(256) void cvt_bf16(const float* __restrict__ in,
                                                unsigned short* __restrict__ out, int n) {
  int i = (blockIdx.x * blockDim.x + threadIdx.x) * 4;
  if (i >= n) return;
  float4 f = *(const float4*)(in + i);
  uint2 p;
  p.x = (unsigned int)f2bf(f.x) | ((unsigned int)f2bf(f.y) << 16);
  p.y = (unsigned int)f2bf(f.z) | ((unsigned int)f2bf(f.w) << 16);
  *(uint2*)(out + i) = p;
}

// ---------------- WMMA GEMM: Out = A[M,K] * W[N,K]^T ----------------
// Each wave: 32x64 output tile (2 A-frag row groups x 4 B-tiles, 8 WMMA / k-step).
// Block = 8 waves covering 128x128.
// MODE 0: bf16 row-major out   MODE 1: bf16 transposed out [(b*HID+col)*SQ+s]
// MODE 2: f32 row-major out
template <int MODE>
__global__ __launch_bounds__(256) void gemm_xwT(const unsigned short* __restrict__ A,
                                                const unsigned short* __restrict__ W,
                                                void* __restrict__ Out,
                                                int M, int N, int K) {
  const int lane = threadIdx.x & 31;
  const int wv   = threadIdx.x >> 5;
  const int hl   = lane >> 4;   // lane-half: selects K sub-range per ISA layout
  const int ln   = lane & 15;
  const int wm   = wv & 3;
  const int wn   = wv >> 2;
  const int n0   = blockIdx.x * 128 + wn * 64;
  const int m0   = blockIdx.y * 128 + wm * 32;

  v8f acc[2][4];
#pragma unroll
  for (int g = 0; g < 2; ++g)
#pragma unroll
    for (int t = 0; t < 4; ++t)
#pragma unroll
      for (int i = 0; i < 8; ++i) acc[g][t][i] = 0.f;

  const unsigned short* arow0 = A + (size_t)(m0 + ln) * K;
  const unsigned short* arow1 = A + (size_t)(m0 + 16 + ln) * K;
  for (int k0 = 0; k0 < K; k0 += 32) {
    FragB a0, a1;                              // A: rows of x, K-chunks {hl*8, 16+hl*8}
    a0.q[0] = *(const uint4*)(arow0 + k0 + hl * 8);
    a0.q[1] = *(const uint4*)(arow0 + k0 + 16 + hl * 8);
    a1.q[0] = *(const uint4*)(arow1 + k0 + hl * 8);
    a1.q[1] = *(const uint4*)(arow1 + k0 + 16 + hl * 8);
    __builtin_prefetch(arow0 + k0 + 128, 0, 3);   // global_prefetch_b8: next k-tiles
    __builtin_prefetch(arow1 + k0 + 128, 0, 3);
#pragma unroll
    for (int t = 0; t < 4; ++t) {              // B: rows of W (x*W^T => contiguous in K)
      FragB bb;
      const unsigned short* br = W + (size_t)(n0 + t * 16 + ln) * K + k0 + hl * 8;
      bb.q[0] = *(const uint4*)(br);
      bb.q[1] = *(const uint4*)(br + 16);
      acc[0][t] = __builtin_amdgcn_wmma_f32_16x16x32_bf16(false, a0.v, false, bb.v,
                                                          (short)0, acc[0][t], false, false);
      acc[1][t] = __builtin_amdgcn_wmma_f32_16x16x32_bf16(false, a1.v, false, bb.v,
                                                          (short)0, acc[1][t], false, false);
    }
  }

  if (MODE == 0) {
    unsigned short* O = (unsigned short*)Out;
#pragma unroll
    for (int g = 0; g < 2; ++g)
#pragma unroll
      for (int t = 0; t < 4; ++t)
#pragma unroll
        for (int v = 0; v < 8; ++v)
          O[(size_t)(m0 + g * 16 + hl * 8 + v) * N + n0 + t * 16 + ln] = f2bf(acc[g][t][v]);
  } else if (MODE == 1) {
    unsigned short* O = (unsigned short*)Out;
#pragma unroll
    for (int g = 0; g < 2; ++g) {
      const int row0 = m0 + g * 16 + hl * 8;
      const int b = row0 / SQ, s = row0 % SQ;
#pragma unroll
      for (int t = 0; t < 4; ++t) {
        const int col = n0 + t * 16 + ln;
        unsigned short tmp[8];
#pragma unroll
        for (int v = 0; v < 8; ++v) tmp[v] = f2bf(acc[g][t][v]);
        *(uint4*)(O + (size_t)(b * HID + col) * SQ + s) = *(const uint4*)tmp;
      }
    }
  } else {
    float* O = (float*)Out;
#pragma unroll
    for (int g = 0; g < 2; ++g)
#pragma unroll
      for (int t = 0; t < 4; ++t)
#pragma unroll
        for (int v = 0; v < 8; ++v)
          O[(size_t)(m0 + g * 16 + hl * 8 + v) * N + n0 + t * 16 + ln] = acc[g][t][v];
  }
}

// ---------------- flash attention: one wave per 32-query tile ----------------
// K fragments loaded once per 32-key block and reused by both 16-row query
// groups; each V fragment feeds two PV WMMAs (16 WMMA per 16 b128 loads).
__global__ __launch_bounds__(256) void flash_attn(const unsigned short* __restrict__ Q,
                                                  const unsigned short* __restrict__ Kk,
                                                  const unsigned short* __restrict__ Vt,
                                                  const unsigned char* __restrict__ mask,
                                                  unsigned short* __restrict__ Oc) {
  __shared__ unsigned short lds[8][32 * 32];   // wave-private P tiles (bf16), 2KB/wave
  const int lane = threadIdx.x & 31;
  const int wv   = threadIdx.x >> 5;
  const int hl   = lane >> 4;
  const int ln   = lane & 15;

  const int widx   = blockIdx.x * 8 + wv;
  const int qtiles = SQ / 32;
  const int q0 = (widx % qtiles) * 32;
  const int h  = (widx / qtiles) % NH;
  const int b  = widx / (qtiles * NH);

  // Q A-fragments: [group (rows +0/+16)][d 0..31 / 32..63]
  FragB qa[2][2];
#pragma unroll
  for (int g = 0; g < 2; ++g) {
    const unsigned short* qrow = Q + (size_t)(b * SQ + q0 + g * 16 + ln) * HID + h * HD;
    qa[g][0].q[0] = *(const uint4*)(qrow + hl * 16);
    qa[g][0].q[1] = *(const uint4*)(qrow + 8 + hl * 16);
    qa[g][1].q[0] = *(const uint4*)(qrow + 32 + hl * 16);
    qa[g][1].q[1] = *(const uint4*)(qrow + 40 + hl * 16);
  }

  v8f o[2][4];
  float mrow[2][8], lrow[2][8];
#pragma unroll
  for (int g = 0; g < 2; ++g) {
#pragma unroll
    for (int t = 0; t < 4; ++t)
#pragma unroll
      for (int i = 0; i < 8; ++i) o[g][t][i] = 0.f;
#pragma unroll
    for (int v = 0; v < 8; ++v) { mrow[g][v] = -3.0e38f; lrow[g][v] = 0.f; }
  }

  const unsigned short* kbase = Kk + (size_t)b * SQ * HID + h * HD;
  const unsigned short* vbase = Vt + (size_t)(b * HID + h * HD) * SQ;
  const unsigned char*  mbase = mask + (size_t)b * SQ;
  unsigned short* myLds = lds[wv];
  const float scale = 0.125f;  // 1/sqrt(64)

  for (int kb = 0; kb < SQ; kb += 32) {
    // ---- K fragments for this 32-key block (shared by both query groups) ----
    FragB kf[2][2];  // [key sub-block +0/+16][d 0..31 / 32..63]
    {
      const unsigned short* kr0 = kbase + (size_t)(kb + ln) * HID + hl * 16;
      kf[0][0].q[0] = *(const uint4*)(kr0);       kf[0][0].q[1] = *(const uint4*)(kr0 + 8);
      kf[0][1].q[0] = *(const uint4*)(kr0 + 32);  kf[0][1].q[1] = *(const uint4*)(kr0 + 40);
      const unsigned short* kr1 = kbase + (size_t)(kb + 16 + ln) * HID + hl * 16;
      kf[1][0].q[0] = *(const uint4*)(kr1);       kf[1][0].q[1] = *(const uint4*)(kr1 + 8);
      kf[1][1].q[0] = *(const uint4*)(kr1 + 32);  kf[1][1].q[1] = *(const uint4*)(kr1 + 40);
    }

    // key-column mask (same column for both lane-halves in C layout)
    const float mk0 = mbase[kb + ln]      ? 0.f : -1.0e10f;
    const float mk1 = mbase[kb + 16 + ln] ? 0.f : -1.0e10f;

#pragma unroll
    for (int g = 0; g < 2; ++g) {
      // ---- scores for query group g ----
      v8f s0, s1;
#pragma unroll
      for (int i = 0; i < 8; ++i) { s0[i] = 0.f; s1[i] = 0.f; }
      s0 = __builtin_amdgcn_wmma_f32_16x16x32_bf16(false, qa[g][0].v, false, kf[0][0].v, (short)0, s0, false, false);
      s0 = __builtin_amdgcn_wmma_f32_16x16x32_bf16(false, qa[g][1].v, false, kf[0][1].v, (short)0, s0, false, false);
      s1 = __builtin_amdgcn_wmma_f32_16x16x32_bf16(false, qa[g][0].v, false, kf[1][0].v, (short)0, s1, false, false);
      s1 = __builtin_amdgcn_wmma_f32_16x16x32_bf16(false, qa[g][1].v, false, kf[1][1].v, (short)0, s1, false, false);

      // ---- online softmax over this 32-key block ----
      float alpha[8];
#pragma unroll
      for (int v = 0; v < 8; ++v) {
        float a0 = s0[v] * scale + mk0;
        float a1 = s1[v] * scale + mk1;
        float mx = rmax16(fmaxf(a0, a1));
        float mn = fmaxf(mrow[g][v], mx);
        alpha[v] = __expf(mrow[g][v] - mn);
        mrow[g][v] = mn;
        float p0 = __expf(a0 - mn);
        float p1 = __expf(a1 - mn);
        lrow[g][v] = lrow[g][v] * alpha[v] + rsum16(p0 + p1);
        const int r = g * 16 + v + hl * 8;    // C-layout row -> LDS row
        myLds[r * 32 + ln]      = f2bf(p0);
        myLds[r * 32 + 16 + ln] = f2bf(p1);
      }
#pragma unroll
      for (int t = 0; t < 4; ++t)
#pragma unroll
        for (int v = 0; v < 8; ++v) o[g][t][v] *= alpha[v];
    }

    asm volatile("s_wait_dscnt 0" ::: "memory");

    // P as bf16 A-fragments (layout conversion via wave-private LDS)
    FragB pa0, pa1;
    {
      const unsigned short* pr = myLds + ln * 32;
      pa0.q[0] = *(const uint4*)(pr + hl * 8);
      pa0.q[1] = *(const uint4*)(pr + 16 + hl * 8);
      const unsigned short* pr1 = pr + 16 * 32;
      pa1.q[0] = *(const uint4*)(pr1 + hl * 8);
      pa1.q[1] = *(const uint4*)(pr1 + 16 + hl * 8);
    }

    // ---- O += P · V[kb..kb+31]; each V fragment feeds both query groups ----
#pragma unroll
    for (int t = 0; t < 4; ++t) {
      FragB vb;
      const unsigned short* vr = vbase + (size_t)(t * 16 + ln) * SQ + kb + hl * 8;
      vb.q[0] = *(const uint4*)(vr);
      vb.q[1] = *(const uint4*)(vr + 16);
      o[0][t] = __builtin_amdgcn_wmma_f32_16x16x32_bf16(false, pa0.v, false, vb.v,
                                                        (short)0, o[0][t], false, false);
      o[1][t] = __builtin_amdgcn_wmma_f32_16x16x32_bf16(false, pa1.v, false, vb.v,
                                                        (short)0, o[1][t], false, false);
    }
  }

  // ---- normalize and store attended values (bf16, row-major [B*S, HID]) ----
#pragma unroll
  for (int g = 0; g < 2; ++g)
#pragma unroll
    for (int t = 0; t < 4; ++t)
#pragma unroll
      for (int v = 0; v < 8; ++v) {
        const int row = q0 + g * 16 + hl * 8 + v;
        Oc[(size_t)(b * SQ + row) * HID + h * HD + t * 16 + ln] = f2bf(o[g][t][v] / lrow[g][v]);
      }
}

extern "C" void kernel_launch(void* const* d_in, const int* in_sizes, int n_in,
                              void* d_out, int out_size, void* d_ws, size_t ws_size,
                              hipStream_t stream) {
  const float* x  = (const float*)d_in[0];
  const unsigned char* mask = (const unsigned char*)d_in[1];
  const float* Wq = (const float*)d_in[2];
  const float* Wk = (const float*)d_in[3];
  const float* Wv = (const float*)d_in[4];
  const float* Wo = (const float*)d_in[5];

  char* ws = (char*)d_ws;
  const size_t MB = 1024u * 1024u;
  unsigned short* xb  = (unsigned short*)(ws + 0 * MB);   // [4096,1024] bf16
  unsigned short* wqb = (unsigned short*)(ws + 8 * MB);
  unsigned short* wkb = (unsigned short*)(ws + 10 * MB);
  unsigned short* wvb = (unsigned short*)(ws + 12 * MB);
  unsigned short* wob = (unsigned short*)(ws + 14 * MB);
  unsigned short* Qb  = (unsigned short*)(ws + 16 * MB);  // [4096,1024]
  unsigned short* Kb  = (unsigned short*)(ws + 24 * MB);  // [4096,1024]
  unsigned short* Vtb = (unsigned short*)(ws + 32 * MB);  // [B*HID, SQ] (transposed)
  unsigned short* Ab  = (unsigned short*)(ws + 40 * MB);  // attended, [4096,1024]

  const int nx = NB * SQ * HID;   // 4194304
  const int nw = HID * HID;       // 1048576
  cvt_bf16<<<nx / 1024, 256, 0, stream>>>(x,  xb,  nx);
  cvt_bf16<<<nw / 1024, 256, 0, stream>>>(Wq, wqb, nw);
  cvt_bf16<<<nw / 1024, 256, 0, stream>>>(Wk, wkb, nw);
  cvt_bf16<<<nw / 1024, 256, 0, stream>>>(Wv, wvb, nw);
  cvt_bf16<<<nw / 1024, 256, 0, stream>>>(Wo, wob, nw);

  dim3 g(HID / 128, (NB * SQ) / 128);
  gemm_xwT<0><<<g, 256, 0, stream>>>(xb, wqb, Qb,  NB * SQ, HID, HID);
  gemm_xwT<0><<<g, 256, 0, stream>>>(xb, wkb, Kb,  NB * SQ, HID, HID);
  gemm_xwT<1><<<g, 256, 0, stream>>>(xb, wvb, Vtb, NB * SQ, HID, HID);

  flash_attn<<<(NB * NH * (SQ / 32)) / 8, 256, 0, stream>>>(Qb, Kb, Vtb, mask, Ab);

  gemm_xwT<2><<<g, 256, 0, stream>>>(Ab, wob, (float*)d_out, NB * SQ, HID, HID);
}
